// NeuralCA_85710367359155
// MI455X (gfx1250) — compile-verified, matrix operand order checked by reading, not run on hardware
//
#include <hip/hip_runtime.h>
#include <hip/hip_bf16.h>

typedef _Float16 half_t;
typedef __attribute__((ext_vector_type(16))) _Float16 v16h;
typedef __attribute__((ext_vector_type(8)))  _Float16 v8h;
typedef __attribute__((ext_vector_type(8)))  float    v8f;

#define NB   8      // batch
#define NC   16     // channels
#define NH   96
#define NW   96
#define NHID 128
#define TW   16     // tile width  (pixels)
#define THT  8      // tile height (pixels)
#define NSTEPS 16

union AFrag { v16h v; v8h h[2]; };

__device__ __forceinline__ unsigned wang(unsigned s) {
    s = (s ^ 61u) ^ (s >> 16); s *= 9u; s ^= s >> 4;
    s *= 0x27d4eb2du; s ^= s >> 15; return s;
}

// ---------------------------------------------------------------------------
// Prep: convert weights to f16 once, zero-padded to the WMMA K layout.
//   w1h: [128][64] f16  (K=48..63 zero)   w2h: [16][128] f16
// ---------------------------------------------------------------------------
__global__ __launch_bounds__(256)
void prep_weights(const float* __restrict__ w1, const float* __restrict__ w2,
                  half_t* __restrict__ w1h, half_t* __restrict__ w2h)
{
    int i = blockIdx.x * 256 + threadIdx.x;
    if (i < NHID * 64) {
        int n = i >> 6, k = i & 63;
        w1h[i] = (k < 48) ? (half_t)w1[n * 48 + k] : (half_t)0.0f;
    }
    if (i < NC * NHID) {
        w2h[i] = (half_t)w2[i];
    }
}

// ---------------------------------------------------------------------------
// Kernel A: one CA inner step (perception -> relu MLP -> stochastic update).
// Writes un-clipped xn to xout. One block = 16x8 pixel tile of one image.
// 256 threads = 8 waves; wave w owns pixel row w (M = 16 pixels for WMMA).
// ---------------------------------------------------------------------------
__global__ __launch_bounds__(256)
void ca_step_kernel(const float* __restrict__ xin,
                    const half_t* __restrict__ w1h, const float* __restrict__ b1,
                    const half_t* __restrict__ w2h, const float* __restrict__ b2,
                    float* __restrict__ xout, int step)
{
    __shared__ __align__(16) float  lx[NC][THT + 2][TW + 4];   // x tile + halo (padded stride)
    __shared__ __align__(16) half_t lp[THT * TW][48];          // perception rows (K=48)
    __shared__ __align__(16) half_t ly[8][16][32];             // per-wave y re-layout scratch
    __shared__ __align__(16) float  lxn[NC][THT][TW];          // staged output tile

    const int tid = threadIdx.x;
    const int bx = blockIdx.x, by = blockIdx.y, b = blockIdx.z;
    const int x0 = bx * TW, y0 = by * THT;

    // ---- stage x tile + 1-pixel halo (zero pad outside image) ----
    for (int i = tid; i < NC * (THT + 2) * (TW + 2); i += 256) {
        int c   = i / ((THT + 2) * (TW + 2));
        int rem = i % ((THT + 2) * (TW + 2));
        int yy  = rem / (TW + 2), xx = rem % (TW + 2);
        int gy = y0 + yy - 1, gx = x0 + xx - 1;
        float v = 0.0f;
        if (gy >= 0 && gy < NH && gx >= 0 && gx < NW)
            v = xin[((b * NC + c) * NH + gy) * NW + gx];
        lx[c][yy][xx] = v;
    }
    __syncthreads();

    // ---- perception: p = [x, sobel_x(x), sobel_y(x)] as f16 rows of 48 ----
    {
        int pix = tid & 127;
        int py = pix >> 4, px = pix & 15;
        int c0 = (tid >> 7) * 8;
        for (int c = c0; c < c0 + 8; ++c) {
            float a00 = lx[c][py    ][px], a01 = lx[c][py    ][px + 1], a02 = lx[c][py    ][px + 2];
            float a10 = lx[c][py + 1][px],                              a12 = lx[c][py + 1][px + 2];
            float a20 = lx[c][py + 2][px], a21 = lx[c][py + 2][px + 1], a22 = lx[c][py + 2][px + 2];
            float gxv = ((a02 - a00) + 2.0f * (a12 - a10) + (a22 - a20)) * 0.125f;
            float gyv = ((a20 - a00) + 2.0f * (a21 - a01) + (a22 - a02)) * 0.125f;
            lp[pix][c]      = (half_t)lx[c][py + 1][px + 1];
            lp[pix][16 + c] = (half_t)gxv;
            lp[pix][32 + c] = (half_t)gyv;
        }
    }
    __syncthreads();

    const int wave = tid >> 5;          // pixel row of this wave
    const int lane = tid & 31;
    const int m    = lane & 15;         // A-fragment row / B-fragment column
    const int hw   = lane >> 4;         // half-wave select (ISA 16-bit layout)

    // ---- A fragments for GEMM1 (perception, K=48 padded to 64) ----
    const half_t* prow = lp[wave * TW + m];
    AFrag a0, a1;
    a0.h[0] = *(const v8h*)(prow + hw * 8);          // K = hw*8 + j       (j<8)
    a0.h[1] = *(const v8h*)(prow + 16 + hw * 8);     // K = 16 + hw*8 + j  (j>=8)
    a1.h[0] = *(const v8h*)(prow + 32 + hw * 8);     // K = 32 + hw*8 + j  (j<8)
    {
        v8h z = {};
        a1.h[1] = z;                                 // K >= 48 : zero pad
    }

    // ---- preload B fragments for GEMM2 (w2^T, K=128, N=16), f16 direct ----
    AFrag b2f[4];
    for (int ki = 0; ki < 4; ++ki) {
        const half_t* wr = w2h + m * NHID + ki * 32 + hw * 8;
        b2f[ki].h[0] = *(const v8h*)(wr);
        b2f[ki].h[1] = *(const v8h*)(wr + 16);
    }

    // ---- fused GEMM1 (48->128, relu) and GEMM2 (128->16) ----
    v8f acc2 = {};
    half_t* lyw = &ly[wave][0][0];

    for (int t = 0; t < 8; ++t) {                    // 8 N-tiles of hidden dim
        const int n = t * 16 + m;
        const half_t* wr = w1h + n * 64;
        AFrag w1f0, w1f1;
        w1f0.h[0] = *(const v8h*)(wr + hw * 8);
        w1f0.h[1] = *(const v8h*)(wr + 16 + hw * 8);
        w1f1.h[0] = *(const v8h*)(wr + 32 + hw * 8);
        w1f1.h[1] = *(const v8h*)(wr + 48 + hw * 8); // pre-zeroed pad

        v8f acc1 = {};
        acc1 = __builtin_amdgcn_wmma_f32_16x16x32_f16(false, a0.v, false, w1f0.v,
                                                      (short)0, acc1, false, false);
        acc1 = __builtin_amdgcn_wmma_f32_16x16x32_f16(false, a1.v, false, w1f1.v,
                                                      (short)0, acc1, false, false);

        // bias + relu, re-layout D (row = r + 8*hw) -> A rows in LDS
        float bias = b1[n];
        int kk = (t & 1) * 16 + m;
        for (int r = 0; r < 8; ++r) {
            float yv = acc1[r] + bias;
            yv = yv > 0.0f ? yv : 0.0f;
            lyw[(r + 8 * hw) * 32 + kk] = (half_t)yv;
        }

        if (t & 1) {                                 // two hidden tiles ready -> one K=32 step
            const half_t* yr = lyw + m * 32;
            AFrag a2;
            a2.h[0] = *(const v8h*)(yr + hw * 8);
            a2.h[1] = *(const v8h*)(yr + 16 + hw * 8);
            acc2 = __builtin_amdgcn_wmma_f32_16x16x32_f16(false, a2.v, false, b2f[t >> 1].v,
                                                          (short)0, acc2, false, false);
        }
    }

    // ---- xn = x + update * fire ; stage tile ----
    {
        const int oc = m;                            // output channel = lane%16
        float bo = b2[oc];
        for (int r = 0; r < 8; ++r) {
            int px = r + 8 * hw;                     // D row -> pixel column
            int gy = y0 + wave, gx = x0 + px;
            unsigned h = wang((((unsigned)step * NB + b) * NH + gy) * NW + gx + 0x9e3779b9u);
            float fire = (h & 1u) ? 1.0f : 0.0f;
            float xc = lx[oc][wave + 1][px + 1];
            lxn[oc][wave][px] = xc + (acc2[r] + bo) * fire;
        }
    }
    __syncthreads();

    // ---- coalesced NCHW store ----
    for (int i = tid; i < NC * THT * TW; i += 256) {
        int c = i >> 7, rem = i & 127;
        int py = rem >> 4, px = rem & 15;
        xout[((b * NC + c) * NH + y0 + py) * NW + x0 + px] = lxn[c][py][px];
    }
}

// ---------------------------------------------------------------------------
// Kernel B: life mask (pre & post 3x3 maxpool on channel 3) + clip.
// ---------------------------------------------------------------------------
__global__ __launch_bounds__(256)
void ca_life_kernel(const float* __restrict__ xold, const float* __restrict__ xnew,
                    float* __restrict__ xfin)
{
    int idx = blockIdx.x * 256 + threadIdx.x;
    if (idx >= NB * NH * NW) return;
    int b = idx / (NH * NW);
    int rem = idx % (NH * NW);
    int y = rem / NW, x = rem % NW;

    float mo = -1e30f, mn = -1e30f;
    const float* po = xold + ((b * NC + 3) * NH) * NW;
    const float* pn = xnew + ((b * NC + 3) * NH) * NW;
    for (int dy = -1; dy <= 1; ++dy) {
        int yy = y + dy;
        if (yy < 0 || yy >= NH) continue;
        for (int dx = -1; dx <= 1; ++dx) {
            int xx = x + dx;
            if (xx < 0 || xx >= NW) continue;
            mo = fmaxf(mo, po[yy * NW + xx]);
            mn = fmaxf(mn, pn[yy * NW + xx]);
        }
    }
    float life = (mo > 0.1f && mn > 0.1f) ? 1.0f : 0.0f;
    for (int c = 0; c < NC; ++c) {
        float v = xnew[((b * NC + c) * NH + y) * NW + x] * life;
        v = fminf(10.0f, fmaxf(-10.0f, v));
        xfin[((b * NC + c) * NH + y) * NW + x] = v;
    }
}

// ---------------------------------------------------------------------------
extern "C" void kernel_launch(void* const* d_in, const int* in_sizes, int n_in,
                              void* d_out, int out_size, void* d_ws, size_t ws_size,
                              hipStream_t stream)
{
    const float* x  = (const float*)d_in[0];
    const float* w1 = (const float*)d_in[1];
    const float* b1 = (const float*)d_in[2];
    const float* w2 = (const float*)d_in[3];
    const float* b2 = (const float*)d_in[4];
    float* out = (float*)d_out;

    const size_t tensor_elems = (size_t)NB * NC * NH * NW;
    float*  buf0 = (float*)d_ws;                    // clipped state between steps
    float*  buf1 = buf0 + tensor_elems;             // un-clipped xn within a step
    half_t* w1h  = (half_t*)(buf1 + tensor_elems);  // 128 x 64 f16 (zero-padded K)
    half_t* w2h  = w1h + NHID * 64;                 // 16 x 128 f16

    prep_weights<<<(NHID * 64 + 255) / 256, 256, 0, stream>>>(w1, w2, w1h, w2h);

    dim3 gridA(NW / TW, NH / THT, NB);              // 6 x 12 x 8 blocks
    dim3 gridB((NB * NH * NW + 255) / 256);

    const float* cur = x;
    for (int s = 0; s < NSTEPS; ++s) {
        ca_step_kernel<<<gridA, 256, 0, stream>>>(cur, w1h, b1, w2h, b2, buf1, s);
        float* dst = (s == NSTEPS - 1) ? out : buf0;
        ca_life_kernel<<<gridB, 256, 0, stream>>>(cur, buf1, dst);
        cur = buf0;
    }
}
